// CSRNet_GFM_7_StyleGAN2_25786983645743
// MI455X (gfx1250) — matrix operand order repaired and checked by reading, not compiled, not run on hardware
//
#include <hip/hip_runtime.h>
#include <hip/hip_bf16.h>

// ---------------------------------------------------------------------------
// CSRNet/StyleGAN2 pipeline for MI455X (gfx1250, wave32, WMMA).
// Activations kept NHWC f16; all 64x64 3x3 convs run as implicit-GEMM with
// v_wmma_f32_16x16x32_f16 (f32 accumulate), bias/lrelu/noise fused in epilogue.
// Weights pre-packed into B-fragment lanes in contiguous k order (k = 16*(lane/16) + elem), same as A.
// ---------------------------------------------------------------------------

typedef __attribute__((ext_vector_type(16))) _Float16 v16h;
typedef __attribute__((ext_vector_type(8)))  _Float16 v8h;
typedef __attribute__((ext_vector_type(8)))  float    v8f;

#define BB 4
#define HH 256
#define WW 256
#define CC 64
// packed weight block: 9 taps * 2 kchunks * 4 ntiles * (32 lanes * 16 halfs)
#define WPACK_HALFS (9*2*4*512)

// ---------------------------------------------------------------------------
// Pack shared conv weights [64][64][3][3] f32 (OIHW) -> B-fragment f16 layout.
// Packed element (t,kc,nt, lane l, slot kk):  w[cout = nt*16 + (l&15)]
//                                             [cin  = kc*32 + (l>>4)*16 + kk][t]
// ---------------------------------------------------------------------------
__global__ __launch_bounds__(256) void pack_shared_k(const float* __restrict__ w,
                                                     _Float16* __restrict__ wp) {
    int idx = blockIdx.x * blockDim.x + threadIdx.x;
    if (idx >= WPACK_HALFS) return;
    int kk   = idx & 15;
    int l    = (idx >> 4) & 31;
    int rest = idx >> 9;
    int nt   = rest & 3;  rest >>= 2;
    int kc   = rest & 1;
    int t    = rest >> 1;
    int g    = l >> 4, n = l & 15;
    int cin  = kc * 32 + g * 16 + kk;
    int cout = nt * 16 + n;
    wp[idx] = (_Float16)w[((size_t)cout * 64 + cin) * 9 + t];
}

// ---------------------------------------------------------------------------
// style[b][co] = emb[b] . mw[co] + mb[co]        (B*64 threads)
// ---------------------------------------------------------------------------
__global__ __launch_bounds__(256) void mod_style_k(const float* __restrict__ emb,
                                                   const float* __restrict__ mw,
                                                   const float* __restrict__ mb,
                                                   float* __restrict__ style) {
    int i = blockIdx.x * blockDim.x + threadIdx.x;
    if (i >= BB * 64) return;
    int b = i >> 6, co = i & 63;
    const float* e  = emb + (size_t)b * 512;
    const float* wv = mw + (size_t)co * 512;
    float a = mb[co];
    for (int k = 0; k < 512; ++k) a += e[k] * wv[k];
    style[i] = a;
}

// ---------------------------------------------------------------------------
// Modulate + demodulate + pack per-sample weights.
// block = (b, cout) -> 4*64 blocks, 64 threads (one per cin).
// ---------------------------------------------------------------------------
__global__ __launch_bounds__(64) void mod_pack_k(const float* __restrict__ base_w, // [1,64,64,3,3]
                                                 const float* __restrict__ style,  // [B,64]
                                                 _Float16* __restrict__ wp) {      // [B][WPACK]
    int b    = blockIdx.x >> 6;
    int cout = blockIdx.x & 63;
    int cin  = threadIdx.x;
    const float scale = 1.0f / 24.0f;   // 1/sqrt(64*9)
    float s = style[b * 64 + cin];
    const float* bw = base_w + ((size_t)cout * 64 + cin) * 9;
    float w9[9], ss = 0.f;
#pragma unroll
    for (int t = 0; t < 9; ++t) { float v = scale * bw[t] * s; w9[t] = v; ss += v * v; }

    __shared__ float red[64];
    red[cin] = ss;
    __syncthreads();
#pragma unroll
    for (int off = 32; off > 0; off >>= 1) {
        if (cin < off) red[cin] += red[cin + off];
        __syncthreads();
    }
    float demod = rsqrtf(red[0] + 1e-8f);

    int kc = cin >> 5, K = cin & 31, g = K >> 4, kk = K & 15;
    int nt = cout >> 4, n = cout & 15, l = g * 16 + n;
#pragma unroll
    for (int t = 0; t < 9; ++t) {
        int off = (((t * 2 + kc) * 4 + nt) * 512) + l * 16 + kk;
        wp[(size_t)b * WPACK_HALFS + off] = (_Float16)(w9[t] * demod);
    }
}

// ---------------------------------------------------------------------------
// First conv: NCHW f32 (3ch) -> NHWC f16 (64ch) + bias + lrelu. Direct (tiny).
// ---------------------------------------------------------------------------
__global__ __launch_bounds__(256) void conv_first_k(const float* __restrict__ x,
                                                    const float* __restrict__ w,   // [64][3][3][3]
                                                    const float* __restrict__ bias,
                                                    _Float16* __restrict__ out) {
    __shared__ float ws[64 * 27];
    __shared__ float bs[64];
    for (int i = threadIdx.x; i < 64 * 27; i += blockDim.x) ws[i] = w[i];
    for (int i = threadIdx.x; i < 64; i += blockDim.x) bs[i] = bias[i];
    __syncthreads();
    int idx = blockIdx.x * blockDim.x + threadIdx.x;
    if (idx >= BB * HH * WW) return;
    int xc = idx & 255, yr = (idx >> 8) & 255, b = idx >> 16;
    float v[27];
#pragma unroll
    for (int c = 0; c < 3; ++c)
#pragma unroll
        for (int t = 0; t < 9; ++t) {
            int yy = yr + t / 3 - 1, xx = xc + t % 3 - 1;
            v[c * 9 + t] = (yy >= 0 && yy < HH && xx >= 0 && xx < WW)
                               ? x[(((size_t)b * 3 + c) * HH + yy) * WW + xx] : 0.f;
        }
    _Float16* op = out + (size_t)idx * CC;
    for (int co = 0; co < 64; ++co) {
        float a = bs[co];
#pragma unroll
        for (int j = 0; j < 27; ++j) a += ws[co * 27 + j] * v[j];
        a = a >= 0.f ? a : 0.1f * a;
        op[co] = (_Float16)a;
    }
}

// ---------------------------------------------------------------------------
// Core 64->64 3x3 conv via WMMA implicit GEMM.
// One wave = 16-pixel strip (along W) x all 64 couts. 72 WMMAs per wave tile.
// ---------------------------------------------------------------------------
template <bool HAS_BIAS, bool LRELU, bool ADD_NOISE, bool PER_SAMPLE_W>
__global__ __launch_bounds__(128) void conv64_wmma(const _Float16* __restrict__ in,   // NHWC f16
                                                   const _Float16* __restrict__ wpack,
                                                   const float* __restrict__ bias,    // [64]
                                                   const float* __restrict__ noise,   // [B,1,H,W]
                                                   const float* __restrict__ wn,      // scalar
                                                   _Float16* __restrict__ out) {      // NHWC f16
    int wave = blockIdx.x * (blockDim.x >> 5) + (threadIdx.x >> 5);
    int xt   = wave & 15;            // 16 tiles per row
    int row  = (wave >> 4) & 255;
    int b    = wave >> 12;
    int x0   = xt * 16;
    int lane = threadIdx.x & 31;
    int g    = lane >> 4;
    int m    = lane & 15;

    const _Float16* wp0 = PER_SAMPLE_W ? wpack + (size_t)b * WPACK_HALFS : wpack;

    v8f acc0 = {}, acc1 = {}, acc2 = {}, acc3 = {};

#pragma unroll
    for (int t = 0; t < 9; ++t) {
        const int dy = t / 3 - 1, dx = t % 3 - 1;
        int yy = row + dy;
        if (yy < 0 || yy >= HH) continue;        // wave-uniform skip
        int xx = x0 + m + dx;
        bool inb = (xx >= 0) && (xx < WW);
        const _Float16* arow =
            in + (((size_t)(b * HH + yy)) * WW + xx) * CC + g * 16;
#pragma unroll
        for (int kc = 0; kc < 2; ++kc) {
            // A fragment: lane group g holds K = g*16+i (elems 0..15)
            v8h z = {};
            v8h lo = z, hi = z;
            if (inb) {
                lo = *(const v8h*)(arow + kc * 32);
                hi = *(const v8h*)(arow + kc * 32 + 8);
            }
            v16h a;
#pragma unroll
            for (int i = 0; i < 8; ++i) { a[i] = lo[i]; a[i + 8] = hi[i]; }

            const _Float16* wb = wp0 + (size_t)((t * 2 + kc) * 4) * 512 + lane * 16;
            v16h b0 = *(const v16h*)(wb);
            v16h b1 = *(const v16h*)(wb + 512);
            v16h b2 = *(const v16h*)(wb + 1024);
            v16h b3 = *(const v16h*)(wb + 1536);

            acc0 = __builtin_amdgcn_wmma_f32_16x16x32_f16(false, a, false, b0, (short)0, acc0, false, false);
            acc1 = __builtin_amdgcn_wmma_f32_16x16x32_f16(false, a, false, b1, (short)0, acc1, false, false);
            acc2 = __builtin_amdgcn_wmma_f32_16x16x32_f16(false, a, false, b2, (short)0, acc2, false, false);
            acc3 = __builtin_amdgcn_wmma_f32_16x16x32_f16(false, a, false, b3, (short)0, acc3, false, false);
        }
    }

    // Epilogue: D layout -> row = r + 8*g, col n = lane&15 within each ntile.
    int n = lane & 15;
    float b0v = HAS_BIAS ? bias[n]      : 0.f;
    float b1v = HAS_BIAS ? bias[16 + n] : 0.f;
    float b2v = HAS_BIAS ? bias[32 + n] : 0.f;
    float b3v = HAS_BIAS ? bias[48 + n] : 0.f;
    float wnv = ADD_NOISE ? wn[0] : 0.f;

#pragma unroll
    for (int r = 0; r < 8; ++r) {
        int mm  = r + g * 8;
        size_t pix = ((size_t)(b * HH + row)) * WW + x0 + mm;
        float nz = ADD_NOISE ? wnv * noise[pix] : 0.f;
        float v0 = acc0[r] + b0v + nz;
        float v1 = acc1[r] + b1v + nz;
        float v2 = acc2[r] + b2v + nz;
        float v3 = acc3[r] + b3v + nz;
        if (LRELU) {
            v0 = v0 >= 0.f ? v0 : 0.1f * v0;
            v1 = v1 >= 0.f ? v1 : 0.1f * v1;
            v2 = v2 >= 0.f ? v2 : 0.1f * v2;
            v3 = v3 >= 0.f ? v3 : 0.1f * v3;
        }
        _Float16* op = out + pix * CC;
        op[n]      = (_Float16)v0;
        op[16 + n] = (_Float16)v1;
        op[32 + n] = (_Float16)v2;
        op[48 + n] = (_Float16)v3;
    }
}

// ---------------------------------------------------------------------------
// Last conv: NHWC f16 (64ch) -> NCHW f32 (3ch) + bias. Direct (tiny).
// ---------------------------------------------------------------------------
__global__ __launch_bounds__(256) void conv_last_k(const _Float16* __restrict__ in,
                                                   const float* __restrict__ w,    // [3][64][3][3]
                                                   const float* __restrict__ bias, // [3]
                                                   float* __restrict__ out) {
    __shared__ float ws[3 * 64 * 9];
    for (int i = threadIdx.x; i < 3 * 64 * 9; i += blockDim.x) ws[i] = w[i];
    __syncthreads();
    int idx = blockIdx.x * blockDim.x + threadIdx.x;
    if (idx >= BB * HH * WW) return;
    int xc = idx & 255, yr = (idx >> 8) & 255, b = idx >> 16;
    float a0 = bias[0], a1 = bias[1], a2 = bias[2];
#pragma unroll
    for (int t = 0; t < 9; ++t) {
        int yy = yr + t / 3 - 1, xx = xc + t % 3 - 1;
        if (yy < 0 || yy >= HH || xx < 0 || xx >= WW) continue;
        const _Float16* p = in + (((size_t)(b * HH + yy)) * WW + xx) * CC;
#pragma unroll
        for (int c = 0; c < 64; ++c) {
            float xv = (float)p[c];
            a0 += ws[(0 * 64 + c) * 9 + t] * xv;
            a1 += ws[(1 * 64 + c) * 9 + t] * xv;
            a2 += ws[(2 * 64 + c) * 9 + t] * xv;
        }
    }
    size_t hw = (size_t)yr * WW + xc;
    out[((size_t)(b * 3 + 0) * HH * WW) + hw] = a0;
    out[((size_t)(b * 3 + 1) * HH * WW) + hw] = a1;
    out[((size_t)(b * 3 + 2) * HH * WW) + hw] = a2;
}

// ---------------------------------------------------------------------------
extern "C" void kernel_launch(void* const* d_in, const int* in_sizes, int n_in,
                              void* d_out, int out_size, void* d_ws, size_t ws_size,
                              hipStream_t stream) {
    (void)in_sizes; (void)n_in; (void)out_size; (void)ws_size;

    const float* x      = (const float*)d_in[0];
    const float* emb    = (const float*)d_in[1];
    const float* noise0 = (const float*)d_in[2];
    const float* noise1 = (const float*)d_in[3];
    const float* noise2 = (const float*)d_in[4];
    const float* w_first = (const float*)d_in[5];
    const float* b_first = (const float*)d_in[6];
    const float* w_hr[5] = {(const float*)d_in[7], (const float*)d_in[9],
                            (const float*)d_in[11], (const float*)d_in[13],
                            (const float*)d_in[15]};
    const float* b_hr[5] = {(const float*)d_in[8], (const float*)d_in[10],
                            (const float*)d_in[12], (const float*)d_in[14],
                            (const float*)d_in[16]};
    const float* w_last = (const float*)d_in[17];
    const float* b_last = (const float*)d_in[18];
    const float* mod_mw[3] = {(const float*)d_in[19], (const float*)d_in[25], (const float*)d_in[31]};
    const float* mod_mb[3] = {(const float*)d_in[20], (const float*)d_in[26], (const float*)d_in[32]};
    const float* mod_w[3]  = {(const float*)d_in[21], (const float*)d_in[27], (const float*)d_in[33]};
    const float* mod_cw[3] = {(const float*)d_in[22], (const float*)d_in[28], (const float*)d_in[34]};
    const float* mod_cb[3] = {(const float*)d_in[23], (const float*)d_in[29], (const float*)d_in[35]};
    const float* mod_wn[3] = {(const float*)d_in[24], (const float*)d_in[30], (const float*)d_in[36]};

    // -------- carve workspace --------
    char* ws = (char*)d_ws;
    auto carve = [&](size_t bytes) -> void* {
        void* p = (void*)ws;
        ws += (bytes + 255) & ~(size_t)255;
        return p;
    };
    const size_t ACT_BYTES = (size_t)BB * HH * WW * CC * sizeof(_Float16); // 33.5 MB
    _Float16* actA = (_Float16*)carve(ACT_BYTES);
    _Float16* actB = (_Float16*)carve(ACT_BYTES);
    // 8 shared packs: cw0, cw1, cw2, hr1..hr5
    _Float16* wpS = (_Float16*)carve((size_t)8 * WPACK_HALFS * sizeof(_Float16));
    // 3 mods x 4 samples
    _Float16* wpM = (_Float16*)carve((size_t)3 * BB * WPACK_HALFS * sizeof(_Float16));
    float* style = (float*)carve((size_t)3 * BB * 64 * sizeof(float));

    _Float16* wp_cw[3] = {wpS + 0 * WPACK_HALFS, wpS + 1 * WPACK_HALFS, wpS + 2 * WPACK_HALFS};
    _Float16* wp_hr[5] = {wpS + 3 * WPACK_HALFS, wpS + 4 * WPACK_HALFS, wpS + 5 * WPACK_HALFS,
                          wpS + 6 * WPACK_HALFS, wpS + 7 * WPACK_HALFS};

    const int PACK_BLKS = (WPACK_HALFS + 255) / 256;   // 144
    const int PIX_BLKS  = (BB * HH * WW) / 256;        // 1024
    const int CONV_BLKS = (BB * HH * WW / 16) / 4;     // 4096 blocks of 4 waves

    // -------- weight packing (tiny, reused by every wave) --------
    for (int i = 0; i < 3; ++i)
        pack_shared_k<<<PACK_BLKS, 256, 0, stream>>>(mod_cw[i], wp_cw[i]);
    for (int i = 0; i < 5; ++i)
        pack_shared_k<<<PACK_BLKS, 256, 0, stream>>>(w_hr[i], wp_hr[i]);
    for (int i = 0; i < 3; ++i) {
        mod_style_k<<<1, 256, 0, stream>>>(emb, mod_mw[i], mod_mb[i], style + i * BB * 64);
        mod_pack_k<<<BB * 64, 64, 0, stream>>>(mod_w[i], style + i * BB * 64,
                                               wpM + (size_t)i * BB * WPACK_HALFS);
    }

    // -------- network --------
    conv_first_k<<<PIX_BLKS, 256, 0, stream>>>(x, w_first, b_first, actA);

    const float* noises[3] = {noise0, noise1, noise2};
    _Float16* cur = actA;
    _Float16* nxt = actB;
    auto swap = [&]() { _Float16* t = cur; cur = nxt; nxt = t; };

    for (int i = 0; i < 3; ++i) {
        // grouped modulated conv (per-sample weights) + wn*noise, no bias, no lrelu
        conv64_wmma<false, false, true, true><<<CONV_BLKS, 128, 0, stream>>>(
            cur, wpM + (size_t)i * BB * WPACK_HALFS, nullptr, noises[i], mod_wn[i], nxt);
        swap();
        // conv_last of mod block: bias + lrelu
        conv64_wmma<true, true, false, false><<<CONV_BLKS, 128, 0, stream>>>(
            cur, wp_cw[i], mod_cb[i], nullptr, nullptr, nxt);
        swap();
        // hr conv i+1 (hr1..hr3): bias, no lrelu
        conv64_wmma<true, false, false, false><<<CONV_BLKS, 128, 0, stream>>>(
            cur, wp_hr[i], b_hr[i], nullptr, nullptr, nxt);
        swap();
    }
    // hr4, hr5
    conv64_wmma<true, false, false, false><<<CONV_BLKS, 128, 0, stream>>>(
        cur, wp_hr[3], b_hr[3], nullptr, nullptr, nxt);
    swap();
    conv64_wmma<true, false, false, false><<<CONV_BLKS, 128, 0, stream>>>(
        cur, wp_hr[4], b_hr[4], nullptr, nullptr, nxt);
    swap();

    conv_last_k<<<PIX_BLKS, 256, 0, stream>>>(cur, w_last, b_last, (float*)d_out);
}
